// ModelA_90417651515511
// MI455X (gfx1250) — matrix operand order repaired and checked by reading, not compiled, not run on hardware
//
#include <hip/hip_runtime.h>
#include <hip/hip_bf16.h>

// ---------------- problem constants ----------------
constexpr int Bc  = 16;
constexpr int Lc  = 1024;
constexpr int Dc  = 256;
constexpr int Hc  = 4;
constexpr int DHc = 64;          // Dc / Hc
constexpr int DFFc = 2048;
constexpr int BLc  = Bc * Lc;    // 16384
constexpr size_t BLDc = (size_t)Bc * Lc * Dc;       // 4,194,304
constexpr size_t BLDFFc = (size_t)Bc * Lc * DFFc;   // 33,554,432

// ---------------- WMMA helpers (gfx1250, wave32) ----------------
typedef __attribute__((ext_vector_type(16))) _Float16 v16h;
typedef __attribute__((ext_vector_type(8)))  _Float16 v8h;
typedef __attribute__((ext_vector_type(8)))  float    v8f;

__device__ __forceinline__ v8f wmma_f32_f16(v16h a, v16h b, v8f c) {
  // (neg_a, A, neg_b, B, c_mod, C, reuse_a, reuse_b)
  return __builtin_amdgcn_wmma_f32_16x16x32_f16(false, a, false, b, (short)0, c,
                                                false, false);
}

// A fragment: 16x32 f16 (MxK), A row-major with leading dim lda.
// ISA layout: lanes 0-15 -> M=0..15 (lanes 16-31 same M), VGPR0..3 halves:
// K = hi*8 + j (j<8), VGPR4..7 halves: K = 16 + hi*8 + (j-8), hi = lane/16.
__device__ __forceinline__ v16h frag_a(const _Float16* A, int lda, int m0,
                                       int k0, int lane) {
  const int m  = m0 + (lane & 15);
  const int hi = lane >> 4;
  const _Float16* p = A + (size_t)m * lda + k0;
  v8h lo = *(const v8h*)(p + hi * 8);
  v8h hh = *(const v8h*)(p + 16 + hi * 8);
  v16h r;
#pragma unroll
  for (int i = 0; i < 8; ++i) { r[i] = lo[i]; r[i + 8] = hh[i]; }
  return r;
}

// B fragment: 32x16 f16 (KxN) loaded from Bt stored N-major (N x K row-major,
// leading dim ldb).  ISA layout: lane n = lane&15, halves j<8 -> K = hi*8 + j,
// j>=8 -> K = 16 + hi*8 + (j-8).
__device__ __forceinline__ v16h frag_b(const _Float16* Bt, int ldb, int n0,
                                       int k0, int lane) {
  const int n  = n0 + (lane & 15);
  const int hi = lane >> 4;
  const _Float16* p = Bt + (size_t)n * ldb + k0;
  v8h lo = *(const v8h*)(p + hi * 8);
  v8h hh = *(const v8h*)(p + 16 + hi * 8);
  v16h r;
#pragma unroll
  for (int i = 0; i < 8; ++i) { r[i] = lo[i]; r[i + 8] = hh[i]; }
  return r;
}

// ---------------- weight transpose + f16 cast ----------------
__global__ void k_w_transpose(const float* __restrict__ W,
                              _Float16* __restrict__ Wt, int K, int N) {
  int idx = blockIdx.x * 256 + threadIdx.x;
  if (idx >= K * N) return;
  int k = idx / N, n = idx % N;
  Wt[(size_t)n * K + k] = (_Float16)W[(size_t)k * N + n];
}

// ---------------- embedding gather ----------------
__global__ void k_embed(const int* __restrict__ si, const int* __restrict__ ci,
                        const float* __restrict__ se, const float* __restrict__ ce,
                        float* __restrict__ x, _Float16* __restrict__ xh) {
  size_t idx = (size_t)blockIdx.x * 256 + threadIdx.x;
  if (idx >= BLDc) return;
  int d  = (int)(idx & (Dc - 1));
  int bl = (int)(idx >> 8);
  float v = se[si[bl] * Dc + d] + ce[ci[bl] * Dc + d];
  x[idx]  = v;
  xh[idx] = (_Float16)v;
}

// ---------------- generic WMMA GEMM ----------------
// out = act( A[MxK]f16 @ Bt[NxK]f16^T + bias[N] + resid[MxN] )
// block = 256 threads = 8 waves; block tile 128(M) x 64(N); wave tile 32x32.
__global__ __launch_bounds__(256)
void k_gemm(const _Float16* __restrict__ A, const _Float16* __restrict__ Bt,
            const float* __restrict__ bias, const float* __restrict__ resid,
            float* __restrict__ outf, _Float16* __restrict__ outh,
            int M, int N, int K, int do_relu) {
  const int lane = threadIdx.x & 31;
  const int w    = threadIdx.x >> 5;
  const int wm   = w & 3, wn = w >> 2;
  const int m0   = blockIdx.y * 128 + wm * 32;
  const int n0   = blockIdx.x * 64  + wn * 32;

  v8f acc[2][2] = {};
  for (int k0 = 0; k0 < K; k0 += 32) {
    v16h a0 = frag_a(A, K, m0,      k0, lane);
    v16h a1 = frag_a(A, K, m0 + 16, k0, lane);
    v16h b0 = frag_b(Bt, K, n0,      k0, lane);
    v16h b1 = frag_b(Bt, K, n0 + 16, k0, lane);
    acc[0][0] = wmma_f32_f16(a0, b0, acc[0][0]);
    acc[0][1] = wmma_f32_f16(a0, b1, acc[0][1]);
    acc[1][0] = wmma_f32_f16(a1, b0, acc[1][0]);
    acc[1][1] = wmma_f32_f16(a1, b1, acc[1][1]);
  }

  const int hi = lane >> 4, ln = lane & 15;
#pragma unroll
  for (int i = 0; i < 2; ++i)
#pragma unroll
    for (int j = 0; j < 2; ++j) {
      int n = n0 + j * 16 + ln;
      float bv = bias[n];
#pragma unroll
      for (int r = 0; r < 8; ++r) {
        int m = m0 + i * 16 + hi * 8 + r;        // C/D layout: M = r + hi*8
        size_t idx = (size_t)m * N + n;
        float v = acc[i][j][r] + bv;
        if (resid) v += resid[idx];
        if (do_relu) v = fmaxf(v, 0.f);
        if (outf) outf[idx] = v;
        if (outh) outh[idx] = (_Float16)v;
      }
    }
}

// ---------------- V transpose: [B,L,H*DH] -> [B,H,DH,L] ----------------
__global__ void k_v_transpose(const _Float16* __restrict__ vh,
                              _Float16* __restrict__ vth) {
  size_t idx = (size_t)blockIdx.x * 256 + threadIdx.x;
  if (idx >= BLDc) return;
  int c  = (int)(idx & (Dc - 1));
  int bl = (int)(idx >> 8);
  int b = bl >> 10, l = bl & (Lc - 1);
  int h = c >> 6,   d = c & (DHc - 1);
  vth[(((size_t)(b * Hc + h)) * DHc + d) * Lc + l] = vh[idx];
}

// ---------------- flash attention (one wave = 16 query rows) ----------------
__global__ __launch_bounds__(256)
void k_flash_attn(const _Float16* __restrict__ qh, const _Float16* __restrict__ kh,
                  const _Float16* __restrict__ vth, _Float16* __restrict__ ctxh) {
  __shared__ __align__(16) _Float16 pbuf[8][16 * 32];   // per-wave P tile

  const int lane = threadIdx.x & 31;
  const int w    = threadIdx.x >> 5;
  const int g    = blockIdx.x * 8 + w;        // 0 .. B*H*(L/16)-1 = 4095
  const int qt   = g & 63;
  const int h    = (g >> 6) & (Hc - 1);
  const int b    = g >> 8;
  const int m0   = qt * 16;
  const int hi   = lane >> 4, ln = lane & 15;

  const _Float16* Qb = qh  + (size_t)b * Lc * Dc + h * DHc;
  const _Float16* Kb = kh  + (size_t)b * Lc * Dc + h * DHc;
  const _Float16* Vt = vth + ((size_t)(b * Hc + h)) * DHc * Lc;

  v16h aq0 = frag_a(Qb, Dc, m0, 0, lane);
  v16h aq1 = frag_a(Qb, Dc, m0, 32, lane);

  v8f o[4] = {};
  float rm[8], rs[8];
#pragma unroll
  for (int r = 0; r < 8; ++r) { rm[r] = -INFINITY; rs[r] = 0.f; }

  for (int kb = 0; kb < Lc / 32; ++kb) {
    const int kk = kb * 32;
    // S = Q @ K^T for 32 keys (two 16-wide tiles), DH=64 -> 2 WMMA each
    v8f s0 = {}, s1 = {};
    s0 = wmma_f32_f16(aq0, frag_b(Kb, Dc, kk, 0, lane), s0);
    s0 = wmma_f32_f16(aq1, frag_b(Kb, Dc, kk, 32, lane), s0);
    s1 = wmma_f32_f16(aq0, frag_b(Kb, Dc, kk + 16, 0, lane), s1);
    s1 = wmma_f32_f16(aq1, frag_b(Kb, Dc, kk + 16, 32, lane), s1);

    // online softmax update (row stats shared across 16 lanes of each half)
#pragma unroll
    for (int r = 0; r < 8; ++r) {
      float a = s0[r] * 0.125f;      // 1/sqrt(64)
      float c = s1[r] * 0.125f;
      float tmax = fmaxf(a, c);
#pragma unroll
      for (int msk = 1; msk < 16; msk <<= 1)
        tmax = fmaxf(tmax, __shfl_xor(tmax, msk, 32));
      float nm    = fmaxf(rm[r], tmax);
      float alpha = __expf(rm[r] - nm);
      float p0 = __expf(a - nm), p1 = __expf(c - nm);
      float psum = p0 + p1;
#pragma unroll
      for (int msk = 1; msk < 16; msk <<= 1)
        psum += __shfl_xor(psum, msk, 32);
      rs[r] = rs[r] * alpha + psum;
      rm[r] = nm;
      o[0][r] *= alpha; o[1][r] *= alpha; o[2][r] *= alpha; o[3][r] *= alpha;
      int row = r + hi * 8;
      pbuf[w][row * 32 + ln]      = (_Float16)p0;   // C-layout -> row-major
      pbuf[w][row * 32 + 16 + ln] = (_Float16)p1;
    }

    // re-pack P (16x32 row-major in LDS) as an A fragment, then P @ V
    v16h ap = frag_a(&pbuf[w][0], 32, 0, 0, lane);
#pragma unroll
    for (int j = 0; j < 4; ++j) {
      v16h bv = frag_b(Vt, Lc, j * 16, kk, lane);   // V^T: contiguous in key
      o[j] = wmma_f32_f16(ap, bv, o[j]);
    }
  }

#pragma unroll
  for (int r = 0; r < 8; ++r) {
    float inv = 1.f / rs[r];
    int m = m0 + hi * 8 + r;
    size_t base = (size_t)(b * Lc + m) * Dc + h * DHc;
#pragma unroll
    for (int j = 0; j < 4; ++j)
      ctxh[base + j * 16 + ln] = (_Float16)(o[j][r] * inv);
  }
}

// ---------------- layernorm (one block per row, D==blockDim==256) -----------
__global__ __launch_bounds__(256)
void k_layernorm(float* __restrict__ y, _Float16* __restrict__ yh,
                 const float* __restrict__ g, const float* __restrict__ be) {
  __shared__ float sred[256];
  const int t = threadIdx.x;
  size_t idx = (size_t)blockIdx.x * Dc + t;
  float v = y[idx];
  sred[t] = v; __syncthreads();
  for (int s = 128; s > 0; s >>= 1) {
    if (t < s) sred[t] += sred[t + s];
    __syncthreads();
  }
  float mean = sred[0] * (1.f / Dc);
  __syncthreads();
  float d = v - mean;
  sred[t] = d * d; __syncthreads();
  for (int s = 128; s > 0; s >>= 1) {
    if (t < s) sred[t] += sred[t + s];
    __syncthreads();
  }
  float var = sred[0] * (1.f / Dc);
  float o = d * rsqrtf(var + 1e-5f) * g[t] + be[t];
  y[idx] = o;
  if (yh) yh[idx] = (_Float16)o;
}

// ---------------- logits (C=3), KL partials, boundary flags -----------------
__global__ __launch_bounds__(256)
void k_logits(const float* __restrict__ y2, const float* __restrict__ Wl,
              const float* __restrict__ bl, int* __restrict__ boundary,
              float* __restrict__ klpart) {
  __shared__ float sred[256];
  const int row = blockIdx.x * 256 + threadIdx.x;
  const float* yr = y2 + (size_t)row * Dc;
  float l0 = bl[0], l1 = bl[1], l2 = bl[2];
  for (int d = 0; d < Dc; ++d) {
    float v = yr[d];
    l0 += v * Wl[d * 3 + 0];
    l1 += v * Wl[d * 3 + 1];
    l2 += v * Wl[d * 3 + 2];
  }
  float mx  = fmaxf(l0, fmaxf(l1, l2));
  float lse = mx + __logf(__expf(l0 - mx) + __expf(l1 - mx) + __expf(l2 - mx));
  float lp0 = l0 - lse, lp1 = l1 - lse, lp2 = l2 - lse;
  const float LP = -1.0986122886681098f;   // log(1/3)
  float kl = LP - (lp0 + lp1 + lp2) * (1.f / 3.f);
  boundary[row] = (l0 >= l1 && l0 >= l2) ? 1 : 0;   // argmax==0 (ties -> 0)
  sred[threadIdx.x] = kl; __syncthreads();
  for (int s = 128; s > 0; s >>= 1) {
    if (threadIdx.x < s) sred[threadIdx.x] += sred[threadIdx.x + s];
    __syncthreads();
  }
  if (threadIdx.x == 0) klpart[blockIdx.x] = sred[0];
}

// ---------------- segment ids + counts (serial per batch: deterministic) ----
__global__ void k_segment(const int* __restrict__ boundary,
                          int* __restrict__ segid, float* __restrict__ counts) {
  const int b = threadIdx.x;
  if (b >= Bc) return;
  for (int l = 0; l < Lc; ++l) counts[b * Lc + l] = 0.f;
  int cnt = 0;
  for (int l = 0; l < Lc; ++l) {
    int closed = (l > 0 && boundary[b * Lc + l]) ? 1 : 0;
    segid[b * Lc + l] = cnt;        // exclusive cumsum of `closed`
    cnt += closed;
  }
  for (int l = 0; l < Lc; ++l) counts[b * Lc + segid[b * Lc + l]] += 1.f;
}

// ---------------- segment sums: thread (b,d) owns column d of batch b -------
__global__ __launch_bounds__(256)
void k_segsum(const float* __restrict__ y2, const int* __restrict__ segid,
              float* __restrict__ sums) {
  const int b = blockIdx.x, d = threadIdx.x;
  for (int l = 0; l < Lc; ++l) sums[(size_t)(b * Lc + l) * Dc + d] = 0.f;
  for (int l = 0; l < Lc; ++l) {
    int s = segid[b * Lc + l];
    sums[(size_t)(b * Lc + s) * Dc + d] += y2[(size_t)(b * Lc + l) * Dc + d];
  }
}

__global__ void k_means(const float* __restrict__ sums,
                        const float* __restrict__ counts,
                        _Float16* __restrict__ meansh) {
  size_t idx = (size_t)blockIdx.x * 256 + threadIdx.x;
  if (idx >= BLDc) return;
  int row = (int)(idx >> 8);
  meansh[idx] = (_Float16)(sums[idx] / fmaxf(counts[row], 1.f));
}

// ---------------- predicate head tail + composition -------------------------
__global__ void k_preds(const _Float16* __restrict__ hiddenh,
                        const float* __restrict__ Wp2, const float* __restrict__ bp2,
                        const float* __restrict__ wc, const float* __restrict__ bc,
                        const float* __restrict__ counts,
                        float* __restrict__ compv, float* __restrict__ validv) {
  const int row = blockIdx.x * 256 + threadIdx.x;
  if (row >= BLc) return;
  float acc = 0.f;
  const _Float16* hr = hiddenh + (size_t)row * Dc;
  for (int d = 0; d < Dc; ++d) acc += (float)hr[d] * Wp2[d];
  float p    = 1.f / (1.f + __expf(-(acc + bp2[0])));
  float comp = p * wc[0] + bc[0];
  float valid = counts[row] > 0.f ? 1.f : 0.f;
  compv[row]  = comp * valid;
  validv[row] = valid;
}

__global__ void k_finalize(const float* __restrict__ compv,
                           const float* __restrict__ validv,
                           const float* __restrict__ klpart,
                           float* __restrict__ out) {
  const int t = threadIdx.x;
  if (t < Bc) {
    float num = 0.f, den = 0.f;
    for (int l = 0; l < Lc; ++l) { num += compv[t * Lc + l]; den += validv[t * Lc + l]; }
    out[t] = 1.f / (1.f + __expf(-(num / den)));
  } else if (t == Bc) {
    float s = 0.f;
    for (int i = 0; i < BLc / 256; ++i) s += klpart[i];
    out[Bc] = s / (float)Bc;
  }
}

// ---------------- host: kernel_launch ----------------
extern "C" void kernel_launch(void* const* d_in, const int* in_sizes, int n_in,
                              void* d_out, int out_size, void* d_ws, size_t ws_size,
                              hipStream_t stream) {
  (void)in_sizes; (void)n_in; (void)out_size; (void)ws_size;

  const int*   si  = (const int*)d_in[0];
  const int*   ci  = (const int*)d_in[1];
  const float* se  = (const float*)d_in[2];
  const float* ce  = (const float*)d_in[3];
  const float* Wq  = (const float*)d_in[4];  const float* bq  = (const float*)d_in[5];
  const float* Wk  = (const float*)d_in[6];  const float* bk  = (const float*)d_in[7];
  const float* Wv  = (const float*)d_in[8];  const float* bv  = (const float*)d_in[9];
  const float* Wo  = (const float*)d_in[10]; const float* bo  = (const float*)d_in[11];
  const float* g1  = (const float*)d_in[12]; const float* be1 = (const float*)d_in[13];
  const float* W1  = (const float*)d_in[14]; const float* bf1 = (const float*)d_in[15];
  const float* W2  = (const float*)d_in[16]; const float* bf2 = (const float*)d_in[17];
  const float* g2  = (const float*)d_in[18]; const float* be2 = (const float*)d_in[19];
  const float* Wl  = (const float*)d_in[20]; const float* bl  = (const float*)d_in[21];
  const float* Wp1 = (const float*)d_in[22]; const float* bp1 = (const float*)d_in[23];
  const float* Wp2 = (const float*)d_in[24]; const float* bp2 = (const float*)d_in[25];
  const float* wc  = (const float*)d_in[26]; const float* bc  = (const float*)d_in[27];

  char* wsp = (char*)d_ws;
  size_t off = 0;
  auto alloc = [&](size_t bytes) -> void* {
    void* p = wsp + off;
    off = (off + bytes + 255) & ~(size_t)255;
    return p;
  };

  float*     x    = (float*)    alloc(BLDc * 4);
  _Float16*  xh   = (_Float16*) alloc(BLDc * 2);
  _Float16*  qh   = (_Float16*) alloc(BLDc * 2);
  _Float16*  kh   = (_Float16*) alloc(BLDc * 2);
  _Float16*  vh   = (_Float16*) alloc(BLDc * 2);
  _Float16*  vth  = (_Float16*) alloc(BLDc * 2);
  _Float16*  ctxh = (_Float16*) alloc(BLDc * 2);
  float*     y    = (float*)    alloc(BLDc * 4);
  _Float16*  yh   = (_Float16*) alloc(BLDc * 2);
  _Float16*  h1h  = (_Float16*) alloc(BLDFFc * 2);
  float*     y2   = (float*)    alloc(BLDc * 4);
  _Float16*  WqT  = (_Float16*) alloc((size_t)Dc * Dc * 2);
  _Float16*  WkT  = (_Float16*) alloc((size_t)Dc * Dc * 2);
  _Float16*  WvT  = (_Float16*) alloc((size_t)Dc * Dc * 2);
  _Float16*  WoT  = (_Float16*) alloc((size_t)Dc * Dc * 2);
  _Float16*  Wp1T = (_Float16*) alloc((size_t)Dc * Dc * 2);
  _Float16*  W1T  = (_Float16*) alloc((size_t)Dc * DFFc * 2);
  _Float16*  W2T  = (_Float16*) alloc((size_t)DFFc * Dc * 2);
  int*       bnd  = (int*)      alloc((size_t)BLc * 4);
  int*       segid= (int*)      alloc((size_t)BLc * 4);
  float*     cnts = (float*)    alloc((size_t)BLc * 4);
  float*     klp  = (float*)    alloc((size_t)(BLc / 256) * 4);
  float*     comv = (float*)    alloc((size_t)BLc * 4);
  float*     valv = (float*)    alloc((size_t)BLc * 4);
  // reuse of dead buffers (all same element counts):
  float*     sums    = x;    // x dead after attn-out residual GEMM
  _Float16*  meansh  = qh;   // q/k dead after attention
  _Float16*  hiddenh = kh;

  const int TPB = 256;
  dim3 gw((int)((Dc * (size_t)Dc + TPB - 1) / TPB));
  hipLaunchKernelGGL(k_w_transpose, gw, dim3(TPB), 0, stream, Wq,  WqT,  Dc, Dc);
  hipLaunchKernelGGL(k_w_transpose, gw, dim3(TPB), 0, stream, Wk,  WkT,  Dc, Dc);
  hipLaunchKernelGGL(k_w_transpose, gw, dim3(TPB), 0, stream, Wv,  WvT,  Dc, Dc);
  hipLaunchKernelGGL(k_w_transpose, gw, dim3(TPB), 0, stream, Wo,  WoT,  Dc, Dc);
  hipLaunchKernelGGL(k_w_transpose, gw, dim3(TPB), 0, stream, Wp1, Wp1T, Dc, Dc);
  dim3 gw1((int)((Dc * (size_t)DFFc + TPB - 1) / TPB));
  hipLaunchKernelGGL(k_w_transpose, gw1, dim3(TPB), 0, stream, W1, W1T, Dc, DFFc);
  hipLaunchKernelGGL(k_w_transpose, gw1, dim3(TPB), 0, stream, W2, W2T, DFFc, Dc);

  dim3 ge((int)((BLDc + TPB - 1) / TPB));
  hipLaunchKernelGGL(k_embed, ge, dim3(TPB), 0, stream, si, ci, se, ce, x, xh);

  // Q, K, V projections (f16 out)
  dim3 gqkv(Dc / 64, BLc / 128);
  hipLaunchKernelGGL(k_gemm, gqkv, dim3(TPB), 0, stream, xh, WqT, bq,
                     (const float*)nullptr, (float*)nullptr, qh, BLc, Dc, Dc, 0);
  hipLaunchKernelGGL(k_gemm, gqkv, dim3(TPB), 0, stream, xh, WkT, bk,
                     (const float*)nullptr, (float*)nullptr, kh, BLc, Dc, Dc, 0);
  hipLaunchKernelGGL(k_gemm, gqkv, dim3(TPB), 0, stream, xh, WvT, bv,
                     (const float*)nullptr, (float*)nullptr, vh, BLc, Dc, Dc, 0);

  hipLaunchKernelGGL(k_v_transpose, ge, dim3(TPB), 0, stream, vh, vth);

  // flash attention: B*H*(L/16)=4096 wave-tiles, 8 waves/block
  hipLaunchKernelGGL(k_flash_attn, dim3(Bc * Hc * (Lc / 16) / 8), dim3(TPB), 0,
                     stream, qh, kh, vth, ctxh);

  // out-proj + residual: y = x + ctx@Wo + bo
  hipLaunchKernelGGL(k_gemm, gqkv, dim3(TPB), 0, stream, ctxh, WoT, bo,
                     x, y, (_Float16*)nullptr, BLc, Dc, Dc, 0);
  hipLaunchKernelGGL(k_layernorm, dim3(BLc), dim3(TPB), 0, stream, y, yh, g1, be1);

  // FFN
  hipLaunchKernelGGL(k_gemm, dim3(DFFc / 64, BLc / 128), dim3(TPB), 0, stream,
                     yh, W1T, bf1, (const float*)nullptr, (float*)nullptr, h1h,
                     BLc, DFFc, Dc, 1);
  hipLaunchKernelGGL(k_gemm, gqkv, dim3(TPB), 0, stream, h1h, W2T, bf2,
                     y, y2, (_Float16*)nullptr, BLc, Dc, DFFc, 0);
  hipLaunchKernelGGL(k_layernorm, dim3(BLc), dim3(TPB), 0, stream, y2,
                     (_Float16*)nullptr, g2, be2);

  // logits / KL / boundaries
  hipLaunchKernelGGL(k_logits, dim3(BLc / 256), dim3(TPB), 0, stream, y2, Wl, bl,
                     bnd, klp);
  hipLaunchKernelGGL(k_segment, dim3(1), dim3(32), 0, stream, bnd, segid, cnts);
  hipLaunchKernelGGL(k_segsum, dim3(Bc), dim3(TPB), 0, stream, y2, segid, sums);
  hipLaunchKernelGGL(k_means, ge, dim3(TPB), 0, stream, sums, cnts, meansh);

  // predicate hidden layer (WMMA) + tail
  hipLaunchKernelGGL(k_gemm, gqkv, dim3(TPB), 0, stream, meansh, Wp1T, bp1,
                     (const float*)nullptr, (float*)nullptr, hiddenh, BLc, Dc, Dc, 1);
  hipLaunchKernelGGL(k_preds, dim3(BLc / 256), dim3(TPB), 0, stream, hiddenh,
                     Wp2, bp2, wc, bc, cnts, comv, valv);
  hipLaunchKernelGGL(k_finalize, dim3(1), dim3(32), 0, stream, comv, valv, klp,
                     (float*)d_out);
}
